// XRNNModel_29789893165593
// MI455X (gfx1250) — matrix operand order repaired and checked by reading, not compiled, not run on hardware
//
#include <hip/hip_runtime.h>
#include <hip/hip_bf16.h>
#include <stdint.h>
#include <stddef.h>

// ---------------------------------------------------------------------------
// Problem constants (fixed by the reference)
// ---------------------------------------------------------------------------
#define B_    32
#define T_    512
#define E_    512
#define K2E_  1024        // 2*E
#define H_    1024
#define G4_   4096        // 4*H
#define V0_   2048
#define V1_   128
#define SKIP_ 4
#define RWG_  64          // persistent workgroups in the recurrence (16 cols each)
#define RCOLS_ 16         // h-columns per recurrence workgroup
#define WPAD_ 1032        // 1024 + 8 bf16 row padding (16B) for bank-friendly LDS

typedef __attribute__((ext_vector_type(16))) __bf16 v16bf;
typedef __attribute__((ext_vector_type(8)))  __bf16 v8bf;
typedef __attribute__((ext_vector_type(8)))  float  v8f;
typedef unsigned int u32x4 __attribute__((ext_vector_type(4)));
typedef int          i32x8 __attribute__((ext_vector_type(8)));
typedef int          i32x4 __attribute__((ext_vector_type(4)));

// ---------------------------------------------------------------------------
// WMMA helpers (CDNA5 v_wmma_f32_16x16x32_bf16, wave32)
// ---------------------------------------------------------------------------
__device__ __forceinline__ v8f wmma_bf16(v16bf a, v16bf b, v8f c) {
  return __builtin_amdgcn_wmma_f32_16x16x32_bf16(false, a, false, b,
                                                 (short)0, c, false, false);
}

// A fragment: 16x32 bf16 tile, row-major source with leading dim `ld`.
__device__ __forceinline__ v16bf load_frag_a(const __bf16* tile, int ld, int lane) {
  const int r  = lane & 15;
  const int kh = (lane & 16) ? 8 : 0;
  const __bf16* row = tile + r * ld + kh;
  v8bf lo = *(const v8bf*)(row);
  v8bf hi = *(const v8bf*)(row + 16);
  v16bf f;
#pragma unroll
  for (int i = 0; i < 8; ++i) { f[i] = lo[i]; f[i + 8] = hi[i]; }
  return f;
}

// B fragment: 32x16 bf16, B[k][n] = W[n][k], W row-major (`ld` = K stride).
__device__ __forceinline__ v16bf load_frag_b(const __bf16* w, int ld, int lane) {
  const int n  = lane & 15;
  const int kh = (lane & 16) ? 8 : 0;
  const __bf16* row = w + n * ld + kh;
  v8bf lo = *(const v8bf*)(row);
  v8bf hi = *(const v8bf*)(row + 16);
  v16bf f;
#pragma unroll
  for (int i = 0; i < 8; ++i) { f[i] = lo[i]; f[i + 8] = hi[i]; }
  return f;
}

__device__ __forceinline__ float sigmoidf_(float x) {
  return 1.0f / (1.0f + __expf(-x));
}

// ---------------------------------------------------------------------------
// CDNA5 async global->LDS copy (ASYNCcnt) + TDM (TENSORcnt)
// ---------------------------------------------------------------------------
__device__ __forceinline__ void async_load_b128(uint32_t lds_off, const __bf16* gptr) {
  asm volatile("global_load_async_to_lds_b128 %0, %1, off"
               :: "v"(lds_off), "v"((uint64_t)(uintptr_t)gptr)
               : "memory");
}
__device__ __forceinline__ void wait_asynccnt_le4() {
  asm volatile("s_wait_asynccnt 0x4" ::: "memory");
}
__device__ __forceinline__ void wait_asynccnt_0() {
  asm volatile("s_wait_asynccnt 0x0" ::: "memory");
}

// Tensor Data Mover: 2D tile load Global->LDS (D# layout per ISA 8.3/8.4).
// data_size_code: 0=1B,1=2B,2=4B.  tile_d0 = contiguous elements per row,
// tile_d1 = rows, d0_stride = tensor row stride (elements).
// This toolchain exposes the 6-arg builtin:
//   (u32x4 g0, i32x8 g1, i32x4 g2, i32x4 g3, i32x8 aux, i32 cpol)
__device__ __forceinline__ void tdm_load_2d(uint32_t lds_addr, const void* gsrc,
                                            uint32_t data_size_code,
                                            uint32_t tile_d0, uint32_t tile_d1,
                                            uint32_t tensor_d0, uint32_t tensor_d1,
                                            uint32_t d0_stride) {
  const uint64_t ga = (uint64_t)(uintptr_t)gsrc;
  u32x4 g0 = { 1u,                                   // count=1, user descriptor
               lds_addr,                             // lds_addr [63:32]
               (uint32_t)ga,                         // global_addr lo
               (uint32_t)(ga >> 32) | (2u << 30) };  // global_addr hi | type=2
  i32x8 g1;
  g1[0] = (int)(data_size_code << 16);               // wg_mask=0 | data_size
  g1[1] = (int)(tensor_d0 << 16);                    // tensor_dim0 (bits 79:48)
  g1[2] = (int)((tensor_d0 >> 16) | (tensor_d1 << 16));
  g1[3] = (int)((tensor_d1 >> 16) | (tile_d0 << 16));
  g1[4] = (int)(tile_d1 & 0xffffu);                  // tile_dim1 | tile_dim2=0
  g1[5] = (int)d0_stride;                            // tensor_dim0_stride lo32
  g1[6] = 0;                                         // stride hi16 | dim1_stride lo16
  g1[7] = 0;
  i32x4 z4 = { 0, 0, 0, 0 };
  i32x8 z8 = { 0, 0, 0, 0, 0, 0, 0, 0 };
  __builtin_amdgcn_tensor_load_to_lds(g0, g1, z4, z4, z8, 0);
}

// ---------------------------------------------------------------------------
// Kernel 0: per-call state init (graph-replay safe / deterministic)
// ---------------------------------------------------------------------------
__global__ void xrnn_init(const float* __restrict__ h0, const float* __restrict__ c0,
                          __bf16* __restrict__ h_buf, float* __restrict__ cws,
                          int* __restrict__ barrier_ctr) {
  int i = blockIdx.x * blockDim.x + threadIdx.x;
  if (i == 0) *barrier_ctr = 0;
  if (i < B_ * H_) {
    cws[i]   = c0[i];
    h_buf[i] = (__bf16)h0[i];
  }
}

// ---------------------------------------------------------------------------
// Kernel 1: fp32 -> bf16 conversion
// ---------------------------------------------------------------------------
__global__ void cvt_f32_bf16(const float* __restrict__ src, __bf16* __restrict__ dst, int n) {
  int i = blockIdx.x * blockDim.x + threadIdx.x;
  if (i < n) dst[i] = (__bf16)src[i];
}

// ---------------------------------------------------------------------------
// Kernel 2: teacher-forced embedding gather -> X[t][b][2E] (bf16)
// ---------------------------------------------------------------------------
__global__ void build_x(const int* __restrict__ tok0, const int* __restrict__ tok1,
                        const float* __restrict__ emb0, const float* __restrict__ emb1,
                        __bf16* __restrict__ X) {
  const int row = blockIdx.x;       // row = t*B + b
  const int t = row >> 5;
  const int b = row & 31;
  const int tu = (t > 0) ? (t - 1) : 0;
  const int a0 = tok0[b * T_ + tu];
  const int a1 = tok1[b * T_ + tu];
  const float* s0 = emb0 + (size_t)a0 * E_;
  const float* s1 = emb1 + (size_t)a1 * E_;
  __bf16* dst = X + (size_t)row * K2E_;
  for (int e = threadIdx.x; e < K2E_; e += blockDim.x) {
    float v = (e < E_) ? s0[e] : s1[e - E_];
    dst[e] = (__bf16)v;
  }
}

// ---------------------------------------------------------------------------
// Kernel 3: LDS-staged, double-buffered WMMA GEMM
//   C[M][N] = A[M][K](bf16) @ W[N][K]^T(bf16) + bias0 + bias1
// 256 threads (8 waves), tile 128x128, BK=32, async global->LDS staging
// (4 async b128 per wave per K-step; s_wait_asynccnt<=4 keeps only the
// next buffer in flight).  LDS rows padded to 40 bf16 for bank spread.
// ---------------------------------------------------------------------------
__global__ void __launch_bounds__(256)
gemm_bf16_wmma(const __bf16* __restrict__ A, const __bf16* __restrict__ W,
               const float* __restrict__ bias0, const float* __restrict__ bias1,
               float* __restrict__ C, int M, int N, int K) {
  __shared__ __bf16 shA[2][128][40];
  __shared__ __bf16 shB[2][128][40];

  const int tid  = threadIdx.x;
  const int lane = tid & 31;
  const int w    = tid >> 5;
  const int wm   = w & 3;              // 4 x 32 = 128 rows
  const int wn   = w >> 2;             // 2 x 64 = 128 cols

  const __bf16* Ablk = A + (size_t)(blockIdx.y * 128) * K;
  const __bf16* Wblk = W + (size_t)(blockIdx.x * 128) * K;

  auto stage = [&](int buf, int k0) {
    int c = tid;                       // chunk: row = c>>2, kc = (c&3)*8
#pragma unroll
    for (int i = 0; i < 2; ++i, c += 256) {
      const int row = c >> 2;
      const int kc  = (c & 3) << 3;
      async_load_b128((uint32_t)(uintptr_t)&shA[buf][row][kc],
                      Ablk + (size_t)row * K + k0 + kc);
      async_load_b128((uint32_t)(uintptr_t)&shB[buf][row][kc],
                      Wblk + (size_t)row * K + k0 + kc);
    }
  };

  v8f acc[2][4] = {};

  stage(0, 0);
  int buf = 0;
  for (int k0 = 0; k0 < K; k0 += 32) {
    const bool more = (k0 + 32) < K;
    if (more) stage(buf ^ 1, k0 + 32);
    if (more) wait_asynccnt_le4(); else wait_asynccnt_0();
    __syncthreads();

    v16bf fa[2], fb[4];
#pragma unroll
    for (int mi = 0; mi < 2; ++mi)
      fa[mi] = load_frag_a(&shA[buf][wm * 32 + mi * 16][0], 40, lane);
#pragma unroll
    for (int ni = 0; ni < 4; ++ni)
      fb[ni] = load_frag_b(&shB[buf][wn * 64 + ni * 16][0], 40, lane);
#pragma unroll
    for (int mi = 0; mi < 2; ++mi)
#pragma unroll
      for (int ni = 0; ni < 4; ++ni)
        acc[mi][ni] = wmma_bf16(fa[mi], fb[ni], acc[mi][ni]);

    __syncthreads();
    buf ^= 1;
  }

  const int mo = (lane & 16) ? 8 : 0;
  const int n  = lane & 15;
  const int mBase = blockIdx.y * 128 + wm * 32;
  const int nBase = blockIdx.x * 128 + wn * 64;
#pragma unroll
  for (int mi = 0; mi < 2; ++mi) {
#pragma unroll
    for (int ni = 0; ni < 4; ++ni) {
      const int nn = nBase + ni * 16 + n;
      float bb = 0.0f;
      if (bias0) bb += bias0[nn];
      if (bias1) bb += bias1[nn];
#pragma unroll
      for (int v = 0; v < 8; ++v) {
        const int m = mBase + mi * 16 + mo + v;
        C[(size_t)m * N + nn] = acc[mi][ni][v] + bb;
      }
    }
  }
}

// ---------------------------------------------------------------------------
// Kernel 4: persistent recurrence. 64 WGs x 128 threads; WG `wg` owns the 16
// h-columns [wg*16, wg*16+16).  Its W_hh slice (4 gates x 16 rows x 1024,
// ~129KB padded) is staged into LDS ONCE via async copies and reused for all
// 512 steps.  Per step, each wave issues a TDM tensor_load_to_lds prefetch of
// its 32x16 Xproj tile (consumed at the epilogue under s_wait_tensorcnt),
// computes its 32x16 gate block with WMMA from LDS-only operands, then the
// LSTM elementwise update.  Device-wide monotonic barrier between steps.
// LDS total: 129KB (W) + 66KB (h_mix) + 8KB (xp) + 8KB (gates) < 320KB WGP LDS.
// ---------------------------------------------------------------------------
__global__ void __launch_bounds__(128)
xrnn_recurrence(const __bf16* __restrict__ Whh_bf,     // [4H][H] bf16
                const float*  __restrict__ Xproj,      // [T][B][4H] fp32 (biases folded)
                const int*    __restrict__ conditions, // [B][T]
                const float*  __restrict__ alpha,      // [1]
                __bf16* __restrict__ h_buf,            // [T+1][B][H] bf16
                __bf16* __restrict__ hs,               // [B][T][H]   bf16
                float*  __restrict__ cws,              // [B][H]      fp32
                float*  __restrict__ out_hsl,          // [B][H] fp32
                float*  __restrict__ out_cfin,         // [B][H] fp32
                int* __restrict__ barrier_ctr) {
  __shared__ __bf16 shW[4][RCOLS_][WPAD_];     // ~129 KB: per-gate W rows
  __shared__ __bf16 sh_hmix[B_][WPAD_];        // ~66 KB
  __shared__ float  sh_xp[4][B_][RCOLS_];      // 8 KB: TDM-prefetched Xproj tiles
  __shared__ float  sh_gates[4][B_][RCOLS_];   // 8 KB
  __shared__ int    sh_idx[B_];

  const int tid  = threadIdx.x;
  const int lane = tid & 31;
  const int w    = tid >> 5;               // wave id == gate chunk q
  const int wg   = blockIdx.x;             // 0..63
  const int c0   = wg * RCOLS_;
  const float sig = sigmoidf_(alpha[0]);

  // ---- one-time W_hh slice staging via async global->LDS ----------------
  // 64 rows (4 gates x 16 cols) x 512 bf16 (k 0..511); 16B chunks.
  for (int c = tid; c < 64 * 64; c += 128) {
    const int row = c >> 6;                // 0..63 ; gate q = row>>4, col r = row&15
    const int kc  = (c & 63) << 3;         // 0..504 step 8
    async_load_b128((uint32_t)(uintptr_t)&shW[row >> 4][row & 15][kc],
                    Whh_bf + (size_t)((row >> 4) * H_ + c0 + (row & 15)) * H_ + kc);
  }
  wait_asynccnt_0();
  __syncthreads();

  for (int t = 0; t < T_; ++t) {
    // TDM prefetch of this wave's Xproj tile (32 rows x 16 f32), overlapped
    // with everything below; consumed at the epilogue.
    tdm_load_2d((uint32_t)(uintptr_t)&sh_xp[w][0][0],
                Xproj + (size_t)t * (B_ * G4_) + (size_t)w * H_ + c0,
                /*data_size=4B*/2u, /*tile_d0=*/RCOLS_, /*tile_d1=*/B_,
                /*tensor_d0=*/G4_, /*tensor_d1=*/1u << 20, /*d0_stride=*/G4_);

    if (tid < B_) {
      int idx = t;
      if (t > SKIP_) {
        int cv = conditions[tid * T_ + t];
        idx = (cv < t) ? cv : t;
      }
      sh_idx[tid] = idx;
    }
    __syncthreads();

    // h_mix = sig*h_last + (1-sig)*h_gather -> LDS (16B vectors)
    const __bf16* hb_t = h_buf + (size_t)t * (B_ * H_);
    for (int u = tid; u < (B_ * H_) / 8; u += 128) {
      const int b = u >> 7;                // 128 chunks per row
      const int h = (u & 127) << 3;
      v8bf hl = *(const v8bf*)(hb_t + b * H_ + h);
      v8bf hg = *(const v8bf*)(h_buf + (size_t)sh_idx[b] * (B_ * H_) + b * H_ + h);
      v8bf o;
#pragma unroll
      for (int i = 0; i < 8; ++i)
        o[i] = (__bf16)(sig * (float)hl[i] + (1.0f - sig) * (float)hg[i]);
      *(v8bf*)(&sh_hmix[b][h]) = o;
    }
    __syncthreads();

    // per-wave 32x16 gate block: h_mix @ W_hh^T, all operands in LDS
    v8f acc0 = {}, acc1 = {};
    for (int k0 = 0; k0 < H_; k0 += 32) {
      v16bf fa0 = load_frag_a(&sh_hmix[0][0]  + k0, WPAD_, lane);
      v16bf fa1 = load_frag_a(&sh_hmix[16][0] + k0, WPAD_, lane);
      v16bf fb  = load_frag_b(&shW[w][0][0] + k0, WPAD_, lane);
      acc0 = wmma_bf16(fa0, fb, acc0);
      acc1 = wmma_bf16(fa1, fb, acc1);
    }

    // epilogue: add TDM-prefetched Xproj tile, stage gate pre-activations
    __builtin_amdgcn_s_wait_tensorcnt(0);
    const int mo = (lane & 16) ? 8 : 0;
    const int n  = lane & 15;
#pragma unroll
    for (int v = 0; v < 8; ++v) {
      const int b0 = mo + v;
      const int b1 = 16 + mo + v;
      sh_gates[w][b0][n] = acc0[v] + sh_xp[w][b0][n];
      sh_gates[w][b1][n] = acc1[v] + sh_xp[w][b1][n];
    }
    __syncthreads();

    // LSTM cell update for this WG's 16 columns
    for (int e = tid; e < B_ * RCOLS_; e += 128) {
      const int b = e >> 4;
      const int j = e & 15;
      const int col = c0 + j;
      const float ig = sigmoidf_(sh_gates[0][b][j]);
      const float fg = sigmoidf_(sh_gates[1][b][j]);
      const float gg = tanhf(sh_gates[2][b][j]);
      const float og = sigmoidf_(sh_gates[3][b][j]);
      const size_t ci = (size_t)b * H_ + col;
      const float cn = fg * cws[ci] + ig * gg;
      cws[ci] = cn;
      const float hn = og * tanhf(cn);
      const __bf16 hb = (__bf16)hn;
      h_buf[(size_t)(t + 1) * (B_ * H_) + ci] = hb;
      hs[((size_t)b * T_ + t) * H_ + col]     = hb;
      if (t == T_ - 1) { out_hsl[ci] = hn; out_cfin[ci] = cn; }
    }

    // device-wide barrier (monotonic counter, sense-free)
    __threadfence();
    __syncthreads();
    if (tid == 0) {
      __hip_atomic_fetch_add(barrier_ctr, 1, __ATOMIC_ACQ_REL, __HIP_MEMORY_SCOPE_AGENT);
      const int target = RWG_ * (t + 1);
      while (__hip_atomic_load(barrier_ctr, __ATOMIC_ACQUIRE, __HIP_MEMORY_SCOPE_AGENT) < target) {
        __builtin_amdgcn_s_sleep(8);
      }
    }
    __syncthreads();
  }
}

// ---------------------------------------------------------------------------
// Host launcher
// ---------------------------------------------------------------------------
extern "C" void kernel_launch(void* const* d_in, const int* in_sizes, int n_in,
                              void* d_out, int out_size, void* d_ws, size_t ws_size,
                              hipStream_t stream) {
  (void)in_sizes; (void)n_in; (void)out_size; (void)ws_size;

  const int*   tokens0 = (const int*)  d_in[0];
  const int*   tokens1 = (const int*)  d_in[1];
  const int*   conds   = (const int*)  d_in[2];
  const float* emb0    = (const float*)d_in[3];
  const float* emb1    = (const float*)d_in[4];
  const float* W_ih    = (const float*)d_in[5];
  const float* W_hh    = (const float*)d_in[6];
  const float* b_ih    = (const float*)d_in[7];
  const float* b_hh    = (const float*)d_in[8];
  const float* dec0_w  = (const float*)d_in[9];
  const float* dec0_b  = (const float*)d_in[10];
  const float* dec1_w  = (const float*)d_in[11];
  const float* dec1_b  = (const float*)d_in[12];
  const float* alpha   = (const float*)d_in[13];
  const float* h0      = (const float*)d_in[14];
  const float* c0      = (const float*)d_in[15];

  float* out = (float*)d_out;
  float* out_dec0 = out;                                          // [B,T,V0]
  float* out_dec1 = out + (size_t)B_ * T_ * V0_;                  // [B,T,V1]
  float* out_hsl  = out_dec1 + (size_t)B_ * T_ * V1_;             // [B,H]
  float* out_cfin = out_hsl + (size_t)B_ * H_;                    // [B,H]

  uint8_t* ws = (uint8_t*)d_ws;
  size_t off = 0;
  auto carve = [&](size_t bytes) -> void* {
    off = (off + 255) & ~(size_t)255;
    void* p = ws + off;
    off += bytes;
    return p;
  };
  __bf16* Xbf     = (__bf16*)carve((size_t)T_ * B_ * K2E_ * 2);
  __bf16* Wih_bf  = (__bf16*)carve((size_t)G4_ * K2E_ * 2);
  __bf16* Whh_bf  = (__bf16*)carve((size_t)G4_ * H_ * 2);
  __bf16* D0w_bf  = (__bf16*)carve((size_t)V0_ * H_ * 2);
  __bf16* D1w_bf  = (__bf16*)carve((size_t)V1_ * H_ * 2);
  __bf16* h_buf   = (__bf16*)carve((size_t)(T_ + 1) * B_ * H_ * 2);
  __bf16* hs_bf   = (__bf16*)carve((size_t)B_ * T_ * H_ * 2);
  float*  cws     = (float*) carve((size_t)B_ * H_ * 4);
  float*  Xproj   = (float*) carve((size_t)T_ * B_ * G4_ * 4);
  int*    barrier = (int*)   carve(256);

  xrnn_init<<<(B_ * H_ + 255) / 256, 256, 0, stream>>>(h0, c0, h_buf, cws, barrier);

  {
    int n;
    n = G4_ * K2E_; cvt_f32_bf16<<<(n + 255) / 256, 256, 0, stream>>>(W_ih, Wih_bf, n);
    n = G4_ * H_;   cvt_f32_bf16<<<(n + 255) / 256, 256, 0, stream>>>(W_hh, Whh_bf, n);
    n = V0_ * H_;   cvt_f32_bf16<<<(n + 255) / 256, 256, 0, stream>>>(dec0_w, D0w_bf, n);
    n = V1_ * H_;   cvt_f32_bf16<<<(n + 255) / 256, 256, 0, stream>>>(dec1_w, D1w_bf, n);
  }

  build_x<<<T_ * B_, 256, 0, stream>>>(tokens0, tokens1, emb0, emb1, Xbf);

  // Xproj = X @ W_ih^T + b_ih + b_hh   (M=16384, N=4096, K=1024)
  gemm_bf16_wmma<<<dim3(G4_ / 128, (T_ * B_) / 128), 256, 0, stream>>>(
      Xbf, Wih_bf, b_ih, b_hh, Xproj, T_ * B_, G4_, K2E_);

  // persistent recurrence (64 WGs, grid-barrier per timestep)
  xrnn_recurrence<<<RWG_, 128, 0, stream>>>(
      Whh_bf, Xproj, conds, alpha, h_buf, hs_bf, cws, out_hsl, out_cfin, barrier);

  // decoders: dec = hs @ dec_w^T + b   (M=16384, N=2048/128, K=1024)
  gemm_bf16_wmma<<<dim3(V0_ / 128, (B_ * T_) / 128), 256, 0, stream>>>(
      hs_bf, D0w_bf, dec0_b, nullptr, out_dec0, B_ * T_, V0_, H_);
  gemm_bf16_wmma<<<dim3(V1_ / 128, (B_ * T_) / 128), 256, 0, stream>>>(
      hs_bf, D1w_bf, dec1_b, nullptr, out_dec1, B_ * T_, V1_, H_);
}
